// MiniMaxText01LightningAttention_9491877724595
// MI455X (gfx1250) — matrix operand order repaired and checked by reading, not compiled, not run on hardware
//
#include <hip/hip_runtime.h>

// ---------------------------------------------------------------------------
// MiniMax Text01 Lightning Attention layer for MI455X (gfx1250, wave32, WMMA)
// B=2, N=2048, D=2048, H=16, HD=128
// Pipeline:
//   0) k_f32_to_bf16   : one-time bf16 conversion of x / w_qkv / w_gate / w_out
//   1) k_qkv_gate_gemm : qkv(bf16) = silu(x @ w_qkv^T), gate(f32) = sigmoid(x @ w_gate^T)
//   2) k_lightning_attn: chunked linear-attention scan (WMMA, chunk=128)
//   3) k_rmsnorm_gate  : gated(bf16) = norm_w * RMSNorm(out) * gate
//   4) k_out_gemm      : final = gated @ w_out^T  (fp32 out)
// Matmuls: v_wmma_f32_16x16x32_bf16. LDS staging: global_load_async_to_lds_b128
// (ASYNCcnt) with double buffering.
// ---------------------------------------------------------------------------

typedef __attribute__((ext_vector_type(16))) __bf16 v16bf;
typedef __attribute__((ext_vector_type(8)))  float  v8f;

#define BB    2
#define NN    2048
#define DD    2048
#define HH    16
#define HDIM  128
#define QKV_W 6144   // 3*H*HD
#define GATE_W 2048
#define MTOT  4096   // B*N
#define EPSV  1e-6f

// fp32 -> bf16 round-to-nearest-even (bit trick)
__device__ __forceinline__ unsigned short f2bf(float x) {
  union { float f; unsigned u; } c; c.f = x;
  unsigned r = c.u + 0x7FFFu + ((c.u >> 16) & 1u);
  return (unsigned short)(r >> 16);
}
__device__ __forceinline__ float bf2f(unsigned short h) {
  union { unsigned u; float f; } c; c.u = ((unsigned)h) << 16; return c.f;
}

union FragU { struct { uint4 lo; uint4 hi; } s; v16bf v; };
union U4H  { uint4 u; unsigned short h[8]; };

// A-operand fragment: 16x32 bf16 from row-major [rows][ldk] LDS.
// lane m<16: row m, K {0..7, 16..23}; lanes 16..31: row m-16, K {8..15, 24..31}.
__device__ __forceinline__ v16bf load_fragA(const unsigned short* base,
                                            int row0, int k0, int ldk, int lane) {
  const unsigned short* p = base + (row0 + (lane & 15)) * ldk + k0 + ((lane >> 4) << 3);
  FragU f;
  f.s.lo = *(const uint4*)(p);
  f.s.hi = *(const uint4*)(p + 16);
  return f.v;
}
// B-operand fragment: 32x16 bf16; source row-major [N][K] (= B^T).
// lane n<16: col n, K {0..7, 16..23}; lanes 16..31: col n-16, K {8..15, 24..31}.
__device__ __forceinline__ v16bf load_fragB(const unsigned short* base,
                                            int row0, int k0, int ldk, int lane) {
  const unsigned short* p = base + (row0 + (lane & 15)) * ldk + k0 + ((lane >> 4) << 3);
  FragU f;
  f.s.lo = *(const uint4*)(p);
  f.s.hi = *(const uint4*)(p + 16);
  return f.v;
}

__device__ __forceinline__ v8f wmma_bf16(v16bf a, v16bf b, v8f c) {
  return __builtin_amdgcn_wmma_f32_16x16x32_bf16(false, a, false, b,
                                                 (short)0, c, false, false);
}

// 32-byte async copy global->LDS (two b128 ops, ASYNCcnt-tracked).
// GVS addressing: mem = saddr + voffset + inst_offset; LDS dst = lds_off + inst_offset.
__device__ __forceinline__ void async_cp32(unsigned lds_off, const void* base,
                                           unsigned gbyte_off) {
  asm volatile(
      "global_load_async_to_lds_b128 %0, %1, %2\n\t"
      "global_load_async_to_lds_b128 %0, %1, %2 offset:16"
      :: "v"(lds_off), "v"(gbyte_off), "s"(base)
      : "memory");
}
__device__ __forceinline__ void wait_async_keep4() {
  asm volatile("s_wait_asynccnt 4" ::: "memory");
}
__device__ __forceinline__ void wait_async_all() {
  asm volatile("s_wait_asynccnt 0" ::: "memory");
}
__device__ __forceinline__ unsigned lds_addr(const void* p) {
  return (unsigned)(unsigned long long)p;   // low 32 bits = LDS offset
}

// ---------------------------------------------------------------------------
// Shared 128x128-tile bf16 GEMM core with double-buffered async staging.
// C_tile = A[m0:, :K] @ W[wrow0:, :K]^T ; 8 waves, wave owns 64x32 (4x2 accums)
// ---------------------------------------------------------------------------
__device__ __forceinline__ void gemm_tile_bf16(
    const unsigned short* __restrict__ A, const unsigned short* __restrict__ W,
    int m0, int wrow0, int K,
    unsigned short (&As)[2][128 * 32], unsigned short (&Bs)[2][128 * 32],
    v8f (&acc)[4][2]) {
  const int tid  = threadIdx.x;
  const int lane = tid & 31;
  const int wave = tid >> 5;
  const int wy = wave >> 2, wx = wave & 3;
  const int ra = tid >> 1;            // staging row 0..127
  const int hb = (tid & 1) * 32;      // byte offset within 64B LDS row

  unsigned ldsA[2], ldsB[2];
  ldsA[0] = lds_addr(&As[0][0]) + ra * 64 + hb;
  ldsA[1] = lds_addr(&As[1][0]) + ra * 64 + hb;
  ldsB[0] = lds_addr(&Bs[0][0]) + ra * 64 + hb;
  ldsB[1] = lds_addr(&Bs[1][0]) + ra * 64 + hb;
  const unsigned aRow = (unsigned)((m0 + ra) * K) * 2u + (unsigned)hb;
  const unsigned bRow = (unsigned)((wrow0 + ra) * K) * 2u + (unsigned)hb;

  const int NT = K / 32;
  // prologue: stage tile 0
  async_cp32(ldsA[0], A, aRow);
  async_cp32(ldsB[0], W, bRow);

  for (int it = 0; it < NT; ++it) {
    const int cur = it & 1;
    if (it + 1 < NT) {
      const int nxt = cur ^ 1;
      const unsigned kb = (unsigned)(it + 1) * 64u;  // 32 halves = 64 bytes
      async_cp32(ldsA[nxt], A, aRow + kb);
      async_cp32(ldsB[nxt], W, bRow + kb);
      wait_async_keep4();    // tile `it` complete; tile `it+1` still in flight
    } else {
      wait_async_all();
    }
    __syncthreads();

    v16bf af[4], bfg[2];
#pragma unroll
    for (int i = 0; i < 4; ++i) af[i]  = load_fragA(&As[cur][0], wy * 64 + i * 16, 0, 32, lane);
#pragma unroll
    for (int j = 0; j < 2; ++j) bfg[j] = load_fragB(&Bs[cur][0], wx * 32 + j * 16, 0, 32, lane);
#pragma unroll
    for (int i = 0; i < 4; ++i)
#pragma unroll
      for (int j = 0; j < 2; ++j)
        acc[i][j] = wmma_bf16(af[i], bfg[j], acc[i][j]);
    __syncthreads();   // frag reads done before next tile's copies land here
  }
}

// ---------------------------------------------------------------------------
// Kernel 0: fp32 -> bf16 conversion (one float4 group per thread)
// ---------------------------------------------------------------------------
__global__ __launch_bounds__(256) void k_f32_to_bf16(
    const float* __restrict__ s, unsigned short* __restrict__ d, int n4) {
  int i = blockIdx.x * 256 + threadIdx.x;
  if (i < n4) {
    float4 f = ((const float4*)s)[i];
    uint2 p;
    p.x = (unsigned)f2bf(f.x) | ((unsigned)f2bf(f.y) << 16);
    p.y = (unsigned)f2bf(f.z) | ((unsigned)f2bf(f.w) << 16);
    ((uint2*)d)[i] = p;
  }
}

// ---------------------------------------------------------------------------
// Kernel 1: fused qkv + gate projection. grid = (8192/128, 4096/128)
// ---------------------------------------------------------------------------
__global__ __launch_bounds__(256) void k_qkv_gate_gemm(
    const unsigned short* __restrict__ xh, const unsigned short* __restrict__ wqkvh,
    const unsigned short* __restrict__ wgateh,
    unsigned short* __restrict__ qkvh, float* __restrict__ gate) {
  __shared__ __align__(16) unsigned short As[2][128 * 32];
  __shared__ __align__(16) unsigned short Bs[2][128 * 32];
  const int n0 = blockIdx.x * 128;
  const int m0 = blockIdx.y * 128;
  const bool is_gate = (n0 >= QKV_W);
  const unsigned short* W = is_gate ? wgateh : wqkvh;
  const int wrow0 = is_gate ? (n0 - QKV_W) : n0;

  const int tid = threadIdx.x, lane = tid & 31, wave = tid >> 5;
  const int wy = wave >> 2, wx = wave & 3;
  const int hi8 = (lane >> 4) << 3, ln = lane & 15;

  v8f zero8 = {0.f, 0.f, 0.f, 0.f, 0.f, 0.f, 0.f, 0.f};
  v8f acc[4][2];
#pragma unroll
  for (int i = 0; i < 4; ++i)
#pragma unroll
    for (int j = 0; j < 2; ++j) acc[i][j] = zero8;

  gemm_tile_bf16(xh, W, m0, wrow0, DD, As, Bs, acc);

#pragma unroll
  for (int i = 0; i < 4; ++i)
#pragma unroll
    for (int j = 0; j < 2; ++j)
#pragma unroll
      for (int r = 0; r < 8; ++r) {
        int row = m0 + wy * 64 + i * 16 + r + hi8;
        int col = n0 + wx * 32 + j * 16 + ln;
        float v = acc[i][j][r];
        float sig = __builtin_amdgcn_rcpf(1.0f + __expf(-v));   // fast sigmoid
        if (is_gate) {
          gate[(size_t)row * GATE_W + (col - QKV_W)] = sig;
        } else {
          qkvh[(size_t)row * QKV_W + col] = f2bf(v * sig);      // silu
        }
      }
}

// ---------------------------------------------------------------------------
// Kernel 2: chunked lightning attention. grid = B*H = 32 blocks.
// Per-wave 64x32 register subtile of the 128x128 state^T persists across chunks.
// ---------------------------------------------------------------------------
__global__ __launch_bounds__(256) void k_lightning_attn(
    const unsigned short* __restrict__ qkvh, const float* __restrict__ past_kv,
    const float* __restrict__ slope_rate,
    float* __restrict__ attn_out, float* __restrict__ kv_out) {
  __shared__ __align__(16) unsigned short Qh[HDIM * HDIM];   // [t][e]
  __shared__ __align__(16) unsigned short KhSh[HDIM * HDIM]; // K [s][e], then S [t][s]
  __shared__ __align__(16) unsigned short KT[HDIM * HDIM];   // (lam^(127-s) K)^T [e][s]
  __shared__ __align__(16) unsigned short VT[HDIM * HDIM];   // V^T [d][s]
  __shared__ __align__(16) unsigned short STh[HDIM * HDIM];  // state^T [d][e]

  const int bh = blockIdx.x;
  const int b = bh >> 4, h = bh & 15;
  const int tid = threadIdx.x, lane = tid & 31, wave = tid >> 5;
  const int wy = wave >> 2, wx = wave & 3;
  const int hi8 = (lane >> 4) << 3, ln = lane & 15;
  const float slope = slope_rate[h];
  const float dec128 = __expf(-slope * 128.0f);

  v8f zero8 = {0.f, 0.f, 0.f, 0.f, 0.f, 0.f, 0.f, 0.f};

  v8f st[4][2];   // state^T: element (m=d, n=e) = kv[e][d]
#pragma unroll
  for (int i = 0; i < 4; ++i)
#pragma unroll
    for (int j = 0; j < 2; ++j)
#pragma unroll
      for (int r = 0; r < 8; ++r) {
        int d = wy * 64 + i * 16 + r + hi8;
        int e = wx * 32 + j * 16 + ln;
        st[i][j][r] = past_kv[((size_t)bh * HDIM + e) * HDIM + d];
      }

  const int rr = tid >> 1;        // staging row 0..127
  const int part = tid & 1;       // which 64-half segment

  const unsigned ldsQ = lds_addr(Qh)   + rr * 256 + part * 128;
  const unsigned ldsK = lds_addr(KhSh) + rr * 256 + part * 128;

  for (int c = 0; c < NN / HDIM; ++c) {
    // ---- phase 0: export state to LDS (bf16)
#pragma unroll
    for (int i = 0; i < 4; ++i)
#pragma unroll
      for (int j = 0; j < 2; ++j)
#pragma unroll
        for (int r = 0; r < 8; ++r) {
          int d = wy * 64 + i * 16 + r + hi8;
          int e = wx * 32 + j * 16 + ln;
          STh[d * HDIM + e] = f2bf(st[i][j][r]);
        }

    // ---- phase 1: stage Q,K via async copy; build KT (decayed) and VT in regs
    {
      const size_t rowbase = (size_t)(b * NN + c * HDIM + rr) * QKV_W + h * 384;
      const unsigned qg = (unsigned)((rowbase + part * 64) * 2);   // bytes
#pragma unroll
      for (int g = 0; g < 4; ++g) async_cp32(ldsQ + g * 32, qkvh, qg + g * 32);
#pragma unroll
      for (int g = 0; g < 4; ++g) async_cp32(ldsK + g * 32, qkvh, qg + 256 + g * 32);

      const uint4* kp = (const uint4*)(qkvh + rowbase + 128 + part * 64);
      const uint4* vp = (const uint4*)(qkvh + rowbase + 256 + part * 64);
      const float kdec = __expf(-slope * (float)(127 - rr));
#pragma unroll
      for (int g = 0; g < 8; ++g) {
        U4H ku, vu; ku.u = kp[g]; vu.u = vp[g];
#pragma unroll
        for (int e = 0; e < 8; ++e) {
          int col = part * 64 + g * 8 + e;
          KT[col * HDIM + rr] = f2bf(bf2f(ku.h[e]) * kdec);
          VT[col * HDIM + rr] = vu.h[e];
        }
      }
      wait_async_all();
    }
    __syncthreads();

    // ---- phase 2: S = Q @ K^T
    v8f s_acc[4][2];
#pragma unroll
    for (int i = 0; i < 4; ++i)
#pragma unroll
      for (int j = 0; j < 2; ++j) s_acc[i][j] = zero8;
#pragma unroll
    for (int kk = 0; kk < 4; ++kk) {
      v16bf af[4], bfg[2];
#pragma unroll
      for (int i = 0; i < 4; ++i) af[i]  = load_fragA(Qh,   wy * 64 + i * 16, kk * 32, HDIM, lane);
#pragma unroll
      for (int j = 0; j < 2; ++j) bfg[j] = load_fragB(KhSh, wx * 32 + j * 16, kk * 32, HDIM, lane);
#pragma unroll
      for (int i = 0; i < 4; ++i)
#pragma unroll
        for (int j = 0; j < 2; ++j) s_acc[i][j] = wmma_bf16(af[i], bfg[j], s_acc[i][j]);
    }
    __syncthreads();

    // ---- phase 3: causal decay mask -> Sh (over Kh); scale Qh by lam^(t+1)
#pragma unroll
    for (int i = 0; i < 4; ++i)
#pragma unroll
      for (int j = 0; j < 2; ++j)
#pragma unroll
        for (int r = 0; r < 8; ++r) {
          int t = wy * 64 + i * 16 + r + hi8;
          int s = wx * 32 + j * 16 + ln;
          int dlt = t - s;
          float wgt = (dlt >= 0) ? __expf(-slope * (float)dlt) : 0.0f;
          KhSh[t * HDIM + s] = f2bf(s_acc[i][j][r] * wgt);
        }
    {
      const float qs = __expf(-slope * (float)(rr + 1));
#pragma unroll
      for (int idx = 0; idx < 64; ++idx) {
        int off = rr * HDIM + part * 64 + idx;
        Qh[off] = f2bf(bf2f(Qh[off]) * qs);
      }
    }
    __syncthreads();

    // ---- phase 4: O = S_masked @ V + (lam^(t+1) Q) @ state
    v8f o_acc[4][2];
#pragma unroll
    for (int i = 0; i < 4; ++i)
#pragma unroll
      for (int j = 0; j < 2; ++j) o_acc[i][j] = zero8;
#pragma unroll
    for (int kk = 0; kk < 4; ++kk) {
      v16bf af[4], bfg[2];
#pragma unroll
      for (int i = 0; i < 4; ++i) af[i]  = load_fragA(KhSh, wy * 64 + i * 16, kk * 32, HDIM, lane);
#pragma unroll
      for (int j = 0; j < 2; ++j) bfg[j] = load_fragB(VT,   wx * 32 + j * 16, kk * 32, HDIM, lane);
#pragma unroll
      for (int i = 0; i < 4; ++i)
#pragma unroll
        for (int j = 0; j < 2; ++j) o_acc[i][j] = wmma_bf16(af[i], bfg[j], o_acc[i][j]);
    }
#pragma unroll
    for (int kk = 0; kk < 4; ++kk) {
      v16bf af[4], bfg[2];
#pragma unroll
      for (int i = 0; i < 4; ++i) af[i]  = load_fragA(Qh,  wy * 64 + i * 16, kk * 32, HDIM, lane);
#pragma unroll
      for (int j = 0; j < 2; ++j) bfg[j] = load_fragB(STh, wx * 32 + j * 16, kk * 32, HDIM, lane);
#pragma unroll
      for (int i = 0; i < 4; ++i)
#pragma unroll
        for (int j = 0; j < 2; ++j) o_acc[i][j] = wmma_bf16(af[i], bfg[j], o_acc[i][j]);
    }
#pragma unroll
    for (int i = 0; i < 4; ++i)
#pragma unroll
      for (int j = 0; j < 2; ++j)
#pragma unroll
        for (int r = 0; r < 8; ++r) {
          int t = wy * 64 + i * 16 + r + hi8;
          int d = wx * 32 + j * 16 + ln;
          attn_out[(size_t)(b * NN + c * HDIM + t) * DD + h * HDIM + d] = o_acc[i][j][r];
        }

    // ---- phase 5: state^T = lam^128 * state^T + V^T @ Kdec
#pragma unroll
    for (int i = 0; i < 4; ++i)
#pragma unroll
      for (int j = 0; j < 2; ++j) st[i][j] = st[i][j] * dec128;
#pragma unroll
    for (int kk = 0; kk < 4; ++kk) {
      v16bf af[4], bfg[2];
#pragma unroll
      for (int i = 0; i < 4; ++i) af[i]  = load_fragA(VT, wy * 64 + i * 16, kk * 32, HDIM, lane);
#pragma unroll
      for (int j = 0; j < 2; ++j) bfg[j] = load_fragB(KT, wx * 32 + j * 16, kk * 32, HDIM, lane);
#pragma unroll
      for (int i = 0; i < 4; ++i)
#pragma unroll
        for (int j = 0; j < 2; ++j) st[i][j] = wmma_bf16(af[i], bfg[j], st[i][j]);
    }
    __syncthreads();
  }

  // final state: kv[e][d] = st(m=d, n=e)
#pragma unroll
  for (int i = 0; i < 4; ++i)
#pragma unroll
    for (int j = 0; j < 2; ++j)
#pragma unroll
      for (int r = 0; r < 8; ++r) {
        int d = wy * 64 + i * 16 + r + hi8;
        int e = wx * 32 + j * 16 + ln;
        kv_out[((size_t)bh * HDIM + e) * HDIM + d] = st[i][j][r];
      }
}

// ---------------------------------------------------------------------------
// Kernel 3: RMSNorm over 2048 + gate multiply -> bf16. grid = 4096.
// ---------------------------------------------------------------------------
__global__ __launch_bounds__(256) void k_rmsnorm_gate(
    const float* __restrict__ attn, const float* __restrict__ gate,
    const float* __restrict__ norm_w, unsigned short* __restrict__ gatedh) {
  __shared__ float red[256];
  const int row = blockIdx.x;
  const int tid = threadIdx.x;
  const float* a = attn + (size_t)row * DD;
  float ss = 0.f;
  for (int j = tid; j < DD; j += 256) { float v = a[j]; ss += v * v; }
  red[tid] = ss;
  __syncthreads();
  for (int s = 128; s > 0; s >>= 1) {
    if (tid < s) red[tid] += red[tid + s];
    __syncthreads();
  }
  float scale = rsqrtf(red[0] / (float)DD + EPSV);
  const float* g = gate + (size_t)row * DD;
  unsigned short* o = gatedh + (size_t)row * DD;
  for (int j = tid; j < DD; j += 256) o[j] = f2bf(norm_w[j] * a[j] * scale * g[j]);
}

// ---------------------------------------------------------------------------
// Kernel 4: final projection. grid = (2048/128, 4096/128)
// ---------------------------------------------------------------------------
__global__ __launch_bounds__(256) void k_out_gemm(
    const unsigned short* __restrict__ Ah, const unsigned short* __restrict__ wouth,
    float* __restrict__ out) {
  __shared__ __align__(16) unsigned short As[2][128 * 32];
  __shared__ __align__(16) unsigned short Bs[2][128 * 32];
  const int n0 = blockIdx.x * 128;
  const int m0 = blockIdx.y * 128;
  const int tid = threadIdx.x, lane = tid & 31, wave = tid >> 5;
  const int wy = wave >> 2, wx = wave & 3;
  const int hi8 = (lane >> 4) << 3, ln = lane & 15;

  v8f zero8 = {0.f, 0.f, 0.f, 0.f, 0.f, 0.f, 0.f, 0.f};
  v8f acc[4][2];
#pragma unroll
  for (int i = 0; i < 4; ++i)
#pragma unroll
    for (int j = 0; j < 2; ++j) acc[i][j] = zero8;

  gemm_tile_bf16(Ah, wouth, m0, n0, DD, As, Bs, acc);

#pragma unroll
  for (int i = 0; i < 4; ++i)
#pragma unroll
    for (int j = 0; j < 2; ++j)
#pragma unroll
      for (int r = 0; r < 8; ++r) {
        int row = m0 + wy * 64 + i * 16 + r + hi8;
        int col = n0 + wx * 32 + j * 16 + ln;
        out[(size_t)row * DD + col] = acc[i][j][r];
      }
}

// ---------------------------------------------------------------------------
extern "C" void kernel_launch(void* const* d_in, const int* in_sizes, int n_in,
                              void* d_out, int out_size, void* d_ws, size_t ws_size,
                              hipStream_t stream) {
  (void)in_sizes; (void)n_in; (void)out_size; (void)ws_size;
  const float* x          = (const float*)d_in[0];   // [B,N,D]
  const float* past_kv    = (const float*)d_in[1];   // [B,H,HD,HD]
  const float* slope_rate = (const float*)d_in[2];   // [H]
  const float* w_qkv      = (const float*)d_in[3];   // [6144,2048]
  const float* w_gate     = (const float*)d_in[4];   // [2048,2048]
  const float* w_out      = (const float*)d_in[5];   // [2048,2048]
  const float* norm_w     = (const float*)d_in[6];   // [2048]

  float* out    = (float*)d_out;                     // [B,N,D]
  float* kv_out = out + (size_t)MTOT * DD;           // [B,H,HD,HD]

  char* w = (char*)d_ws;
  unsigned short* xh     = (unsigned short*)w; w += (size_t)MTOT * DD * 2;
  unsigned short* wqkvh  = (unsigned short*)w; w += (size_t)QKV_W * DD * 2;
  unsigned short* wgateh = (unsigned short*)w; w += (size_t)GATE_W * DD * 2;
  unsigned short* wouth  = (unsigned short*)w; w += (size_t)DD * DD * 2;
  unsigned short* qkvh   = (unsigned short*)w; w += (size_t)MTOT * QKV_W * 2;
  float*          gate   = (float*)w;          w += (size_t)MTOT * GATE_W * 4;
  float*          attn   = (float*)w;          w += (size_t)MTOT * DD * 4;
  unsigned short* gatedh = (unsigned short*)w; w += (size_t)MTOT * DD * 2;

  // bf16 conversions
  {
    int n4;
    n4 = MTOT * DD / 4;
    k_f32_to_bf16<<<(n4 + 255) / 256, 256, 0, stream>>>(x, xh, n4);
    n4 = QKV_W * DD / 4;
    k_f32_to_bf16<<<(n4 + 255) / 256, 256, 0, stream>>>(w_qkv, wqkvh, n4);
    n4 = GATE_W * DD / 4;
    k_f32_to_bf16<<<(n4 + 255) / 256, 256, 0, stream>>>(w_gate, wgateh, n4);
    n4 = DD * DD / 4;
    k_f32_to_bf16<<<(n4 + 255) / 256, 256, 0, stream>>>(w_out, wouth, n4);
  }

  dim3 g1((QKV_W + GATE_W) / 128, MTOT / 128);
  k_qkv_gate_gemm<<<g1, 256, 0, stream>>>(xh, wqkvh, wgateh, qkvh, gate);

  k_lightning_attn<<<BB * HH, 256, 0, stream>>>(qkvh, past_kv, slope_rate, attn, kv_out);

  k_rmsnorm_gate<<<MTOT, 256, 0, stream>>>(attn, gate, norm_w, gatedh);

  dim3 g2(DD / 128, MTOT / 128);
  k_out_gemm<<<g2, 256, 0, stream>>>(gatedh, wouth, out);
}
